// Attention_10565619548303
// MI455X (gfx1250) — matrix operand order repaired and checked by reading, not compiled, not run on hardware
//
#include <hip/hip_runtime.h>
#include <cstdint>
#include <cstddef>

typedef __attribute__((ext_vector_type(16))) __bf16        v16bf;
typedef __attribute__((ext_vector_type(8)))  float         v8f;
typedef __attribute__((ext_vector_type(4)))  unsigned int  v4u;
typedef __attribute__((ext_vector_type(8)))  unsigned int  v8u;

#define B_    4
#define N_    2048
#define DIM_  512
#define H_    8
#define DH_   64
#define BH_   (B_ * H_)
#define QKVN_ (3 * H_ * DH_)   // 1536
#define SCALE_ 0.125f          // 64^-0.5

// ---------- helpers ----------

__device__ inline __bf16 f2bf(float f) {
  // round-to-nearest-even f32 -> bf16 via integer ops
  unsigned u = __builtin_bit_cast(unsigned, f);
  unsigned r = (u + 0x7FFFu + ((u >> 16) & 1u)) >> 16;
  unsigned short s = (unsigned short)r;
  return __builtin_bit_cast(__bf16, s);
}

__device__ inline v8f wmma_bf16(v16bf a, v16bf b, v8f c) {
  // D = A(16x32) * B(32x16) + C(16x16 f32)
  return __builtin_amdgcn_wmma_f32_16x16x32_bf16(
      /*neg_a=*/false, a, /*neg_b=*/false, b,
      /*c_mod=*/(short)0, c, /*reuse_a=*/false, /*reuse_b=*/false);
}

// Async DMA: copy 16B per lane from global to LDS (ASYNCcnt-tracked).
// VDST VGPR holds the LDS byte offset; VADDR the 64-bit global address.
__device__ inline void async_copy16(unsigned lds_off, const void* gp) {
  unsigned long long ga = (unsigned long long)(uintptr_t)gp;
  asm volatile("global_load_async_to_lds_b128 %0, %1, off"
               :: "v"(lds_off), "v"(ga)
               : "memory");
}

// Load a 16x32 bf16 A-tile, row-major, leading dim lda (elements).
// A layout (wave32): half h = lane/16, row M = lane%16.
//   VGPR j(0..7): K = h*16 + 2j,2j+1 (contiguous, matches B)
__device__ inline v16bf load_a_tile(const __bf16* a, int lda) {
  const int lane = threadIdx.x & 31;
  const int half = lane >> 4;
  const int row  = lane & 15;
  const __bf16* p = a + (size_t)row * lda + half * 16;
  v8u raw = *(const v8u*)p;         // K = half*16 .. +15
  return __builtin_bit_cast(v16bf, raw);
}

// Load a 32x16 bf16 B-tile from B^T storage: bt[n][k] row-major, ldb (elements).
// B layout: lane n = lane%16, kbase = 16*(lane/16); VGPR v holds
// B[kbase+2v][n], B[kbase+2v+1][n] -> 16 contiguous bf16 from bt row n.
__device__ inline v16bf load_bt_tile(const __bf16* bt, int ldb) {
  const int lane = threadIdx.x & 31;
  const __bf16* p = bt + (size_t)(lane & 15) * ldb + ((lane >> 4) << 4);
  v8u raw = *(const v8u*)p;  // 32 contiguous bytes
  return __builtin_bit_cast(v16bf, raw);
}

// ---------- precision-conversion kernels ----------

__global__ void cvt_bf16_kernel(const float* __restrict__ src,
                                __bf16* __restrict__ dst, int n) {
  int i = blockIdx.x * blockDim.x + threadIdx.x;
  if (i < n) dst[i] = f2bf(src[i]);
}

// w[K][Nc] f32 -> wt[Nc][K] bf16 (transpose so GEMM B-tiles load contiguously)
__global__ void cvt_transpose_kernel(const float* __restrict__ w,
                                     __bf16* __restrict__ wt, int K, int Nc) {
  int i = blockIdx.x * blockDim.x + threadIdx.x;
  if (i < K * Nc) {
    int k = i / Nc;
    int n = i - k * Nc;
    wt[(size_t)n * K + k] = f2bf(w[i]);
  }
}

// ---------- QKV projection: [8192 x 512] @ [512 x 1536] ----------
// 4 waves/block (2x2), each wave a 32x64 tile -> block 64x128.
// Q stored pre-scaled by 1/sqrt(dh).
// Q,K -> [BH][N][DH] bf16 ; V -> transposed [BH][DH][N] bf16.

__global__ void __launch_bounds__(128) qkv_gemm_kernel(
    const __bf16* __restrict__ xb, const __bf16* __restrict__ wqkvT,
    __bf16* __restrict__ qd, __bf16* __restrict__ kd, __bf16* __restrict__ vtd) {
  const int lane  = threadIdx.x & 31;
  const int wid   = threadIdx.x >> 5;
  const int half  = lane >> 4;
  const int lanen = lane & 15;
  const int row0 = blockIdx.y * 64 + (wid >> 1) * 32;
  const int col0 = blockIdx.x * 128 + (wid & 1) * 64;

  v8f c[2][4];
#pragma unroll
  for (int i = 0; i < 2; ++i)
#pragma unroll
    for (int j = 0; j < 4; ++j)
#pragma unroll
      for (int r = 0; r < 8; ++r) c[i][j][r] = 0.f;

  for (int k0 = 0; k0 < DIM_; k0 += 32) {
    if (k0 + 64 < DIM_) {
      __builtin_prefetch(xb + (size_t)row0 * DIM_ + k0 + 64, 0, 3);
      __builtin_prefetch(wqkvT + (size_t)col0 * DIM_ + k0 + 64, 0, 3);
    }
    v16bf a0 = load_a_tile(xb + (size_t)row0 * DIM_ + k0, DIM_);
    v16bf a1 = load_a_tile(xb + (size_t)(row0 + 16) * DIM_ + k0, DIM_);
#pragma unroll
    for (int j = 0; j < 4; ++j) {
      v16bf bt = load_bt_tile(wqkvT + (size_t)(col0 + 16 * j) * DIM_ + k0, DIM_);
      c[0][j] = wmma_bf16(a0, bt, c[0][j]);
      c[1][j] = wmma_bf16(a1, bt, c[1][j]);
    }
  }

#pragma unroll
  for (int i = 0; i < 2; ++i)
#pragma unroll
    for (int j = 0; j < 4; ++j)
#pragma unroll
      for (int r = 0; r < 8; ++r) {
        int mrow = row0 + 16 * i + r + 8 * half;   // C layout: M = r + 8*half
        int ncol = col0 + 16 * j + lanen;          // N = lane%16
        int sel = ncol >> 9;        // 0:Q 1:K 2:V
        int rem = ncol & 511;
        int h = rem >> 6, d = rem & 63;
        int b = mrow >> 11, ns = mrow & 2047;
        int bh = (b << 3) + h;
        float f = c[i][j][r];
        if (sel == 0)      qd [((size_t)bh * N_ + ns) * DH_ + d] = f2bf(f * SCALE_);
        else if (sel == 1) kd [((size_t)bh * N_ + ns) * DH_ + d] = f2bf(f);
        else               vtd[((size_t)bh * DH_ + d) * N_ + ns] = f2bf(f);
      }
}

// ---------- flash attention ----------
// grid (N/128, BH), 128 threads = 4 waves; each wave owns 32 query rows.
// K/V tiles are staged cooperatively into LDS with async global->LDS DMA
// (double-buffered, ASYNCcnt-tracked); S never touches HBM.

__global__ void __launch_bounds__(128) flash_attn_kernel(
    const __bf16* __restrict__ q, const __bf16* __restrict__ k,
    const __bf16* __restrict__ vt, const unsigned char* __restrict__ mask,
    __bf16* __restrict__ ao) {
  __shared__ __align__(16) __bf16 kbuf[2][32 * 64];  // keys: [j 0..31][d 0..63]
  __shared__ __align__(16) __bf16 vbuf[2][64 * 32];  // vals: [d 0..63][j 0..31]
  __shared__ __align__(16) __bf16 plds[4][32 * 32];  // per-wave P scratch

  const int tid   = threadIdx.x;
  const int lane  = tid & 31;
  const int wid   = tid >> 5;
  const int half  = lane >> 4;
  const int lanen = lane & 15;
  const int bh = blockIdx.y;
  const int b = bh >> 3, h = bh & 7;
  const int qi = blockIdx.x * 128 + wid * 32;

  const __bf16* kbase = k  + (size_t)bh * N_ * DH_;
  const __bf16* vbase = vt + (size_t)bh * DH_ * N_;

  // Stage one 32-key block (K: 4KB contiguous, V: 64 rows x 64B) into LDS.
  // 256 chunks of 16B each per tile; 2 chunks per thread per tile ->
  // 4 async instructions per wave per stage.
  auto stage = [&](int buf, int j0) {
    const int c0 = tid * 2;
    const int c1 = c0 + 1;
    const char* kg = (const char*)(kbase + (size_t)j0 * DH_);
    unsigned kl = (unsigned)(uintptr_t)&kbuf[buf][0];
    async_copy16(kl + c0 * 16, kg + c0 * 16);
    async_copy16(kl + c1 * 16, kg + c1 * 16);
    unsigned vl = (unsigned)(uintptr_t)&vbuf[buf][0];
    const int r0 = c0 >> 2, s0c = c0 & 3;
    const int r1 = c1 >> 2, s1c = c1 & 3;
    async_copy16(vl + c0 * 16,
                 (const char*)(vbase + (size_t)r0 * N_ + j0) + s0c * 16);
    async_copy16(vl + c1 * 16,
                 (const char*)(vbase + (size_t)r1 * N_ + j0) + s1c * 16);
  };

  // Q A-tiles for 2 row-tiles x 2 d-chunks (pre-scaled by 1/sqrt(dh))
  const __bf16* qb = q + ((size_t)bh * N_ + qi) * DH_;
  v16bf aq[2][2];
#pragma unroll
  for (int i = 0; i < 2; ++i) {
    aq[i][0] = load_a_tile(qb + (size_t)i * 16 * DH_, DH_);
    aq[i][1] = load_a_tile(qb + (size_t)i * 16 * DH_ + 32, DH_);
  }

  v8f o[2][4];
#pragma unroll
  for (int i = 0; i < 2; ++i)
#pragma unroll
    for (int t = 0; t < 4; ++t)
#pragma unroll
      for (int r = 0; r < 8; ++r) o[i][t][r] = 0.f;
  float m[2][8], l[2][8];
#pragma unroll
  for (int i = 0; i < 2; ++i)
#pragma unroll
    for (int r = 0; r < 8; ++r) { m[i][r] = -1e30f; l[i][r] = 0.f; }

  stage(0, 0);

  const int NB = N_ / 32;
  for (int jb = 0; jb < NB; ++jb) {
    const int j0 = jb * 32;
    if (jb + 1 < NB) {
      stage((jb + 1) & 1, j0 + 32);
      // async loads complete in order: <=4 outstanding => stage jb done
      asm volatile("s_wait_asynccnt 0x4" ::: "memory");
    } else {
      asm volatile("s_wait_asynccnt 0x0" ::: "memory");
    }
    __syncthreads();

    // ---- S = Q K^T : 32x32 score tile (B-tiles from LDS) ----
    const __bf16* kb = &kbuf[jb & 1][0];
    v16bf bk00 = load_bt_tile(kb, DH_);                 // keys +0..15, d 0..31
    v16bf bk01 = load_bt_tile(kb + 32, DH_);            // keys +0..15, d 32..63
    v16bf bk10 = load_bt_tile(kb + 16 * DH_, DH_);      // keys +16..31
    v16bf bk11 = load_bt_tile(kb + 16 * DH_ + 32, DH_);
    v8f z = {};
    v8f s[2][2];
#pragma unroll
    for (int i = 0; i < 2; ++i) {
      s[i][0] = wmma_bf16(aq[i][1], bk01, wmma_bf16(aq[i][0], bk00, z));
      s[i][1] = wmma_bf16(aq[i][1], bk11, wmma_bf16(aq[i][0], bk10, z));
    }

    // ---- mask + online softmax per row-tile ----
    float alpha[2][8];
#pragma unroll
    for (int i = 0; i < 2; ++i) {
#pragma unroll
      for (int r = 0; r < 8; ++r) {
        int rg = qi + 16 * i + r + 8 * half;
        size_t mb = ((size_t)b * N_ + rg) * N_ + j0 + lanen;
        if (!mask[mb])      s[i][0][r] = -1e30f;
        if (!mask[mb + 16]) s[i][1][r] = -1e30f;
      }
#pragma unroll
      for (int r = 0; r < 8; ++r) {
        float mx = fmaxf(s[i][0][r], s[i][1][r]);
#pragma unroll
        for (int d = 1; d < 16; d <<= 1) mx = fmaxf(mx, __shfl_xor(mx, d, 32));
        float mn = fmaxf(m[i][r], mx);
        float a  = __expf(m[i][r] - mn);
        float p0 = __expf(s[i][0][r] - mn);
        float p1 = __expf(s[i][1][r] - mn);
        s[i][0][r] = p0; s[i][1][r] = p1;
        float rs = p0 + p1;
#pragma unroll
        for (int d = 1; d < 16; d <<= 1) rs += __shfl_xor(rs, d, 32);
        l[i][r] = l[i][r] * a + rs;
        m[i][r] = mn;
        alpha[i][r] = a;
      }
#pragma unroll
      for (int t = 0; t < 4; ++t)
#pragma unroll
        for (int r = 0; r < 8; ++r) o[i][t][r] *= alpha[i][r];
    }

    // ---- route P (32x32) from C-layout to A-layout via per-wave LDS ----
    __bf16* pl = &plds[wid][0];
#pragma unroll
    for (int i = 0; i < 2; ++i)
#pragma unroll
      for (int r = 0; r < 8; ++r) {
        int row = 16 * i + r + 8 * half;
        pl[row * 32 + lanen]      = f2bf(s[i][0][r]);
        pl[row * 32 + 16 + lanen] = f2bf(s[i][1][r]);
      }
    asm volatile("s_wait_dscnt 0x0" ::: "memory");
    v16bf pa0 = load_a_tile(pl, 32);
    v16bf pa1 = load_a_tile(pl + 16 * 32, 32);

    // ---- O += P V (V tile in LDS, [64][32]; bv shared by both row-tiles) ----
    const __bf16* vb = &vbuf[jb & 1][0];
#pragma unroll
    for (int t = 0; t < 4; ++t) {
      v16bf bv = load_bt_tile(vb + t * 16 * 32, 32);
      o[0][t] = wmma_bf16(pa0, bv, o[0][t]);
      o[1][t] = wmma_bf16(pa1, bv, o[1][t]);
    }

    __syncthreads();  // protect K/V buffer reuse by the next prefetch
  }

  // ---- normalize and store attn output as [B][N][H*DH] bf16 ----
#pragma unroll
  for (int i = 0; i < 2; ++i)
#pragma unroll
    for (int t = 0; t < 4; ++t)
#pragma unroll
      for (int r = 0; r < 8; ++r) {
        int rg = qi + 16 * i + r + 8 * half;
        float v = o[i][t][r] / l[i][r];
        ao[((size_t)b * N_ + rg) * (H_ * DH_) + h * DH_ + t * 16 + lanen] =
            f2bf(v);
      }
}

// ---------- output projection: [8192 x 512] @ [512 x 512] + bias -> f32 ----------
// 4 waves/block (2x2), each wave 32x64 -> block 64x128.

__global__ void __launch_bounds__(128) out_gemm_kernel(
    const __bf16* __restrict__ ab, const __bf16* __restrict__ woutT,
    const float* __restrict__ bias, float* __restrict__ out) {
  const int lane  = threadIdx.x & 31;
  const int wid   = threadIdx.x >> 5;
  const int half  = lane >> 4;
  const int lanen = lane & 15;
  const int row0 = blockIdx.y * 64 + (wid >> 1) * 32;
  const int col0 = blockIdx.x * 128 + (wid & 1) * 64;

  v8f c[2][4];
#pragma unroll
  for (int i = 0; i < 2; ++i)
#pragma unroll
    for (int j = 0; j < 4; ++j)
#pragma unroll
      for (int r = 0; r < 8; ++r) c[i][j][r] = 0.f;

  for (int k0 = 0; k0 < DIM_; k0 += 32) {
    if (k0 + 64 < DIM_) {
      __builtin_prefetch(ab + (size_t)row0 * DIM_ + k0 + 64, 0, 3);
      __builtin_prefetch(woutT + (size_t)col0 * DIM_ + k0 + 64, 0, 3);
    }
    v16bf a0 = load_a_tile(ab + (size_t)row0 * DIM_ + k0, DIM_);
    v16bf a1 = load_a_tile(ab + (size_t)(row0 + 16) * DIM_ + k0, DIM_);
#pragma unroll
    for (int j = 0; j < 4; ++j) {
      v16bf bt = load_bt_tile(woutT + (size_t)(col0 + 16 * j) * DIM_ + k0, DIM_);
      c[0][j] = wmma_bf16(a0, bt, c[0][j]);
      c[1][j] = wmma_bf16(a1, bt, c[1][j]);
    }
  }

#pragma unroll
  for (int i = 0; i < 2; ++i)
#pragma unroll
    for (int j = 0; j < 4; ++j)
#pragma unroll
      for (int r = 0; r < 8; ++r) {
        int mrow = row0 + 16 * i + r + 8 * half;
        int ncol = col0 + 16 * j + lanen;
        out[(size_t)mrow * DIM_ + ncol] = c[i][j][r] + bias[ncol];
      }
}

// ---------- launch ----------

extern "C" void kernel_launch(void* const* d_in, const int* in_sizes, int n_in,
                              void* d_out, int out_size, void* d_ws, size_t ws_size,
                              hipStream_t stream) {
  (void)in_sizes; (void)n_in; (void)out_size; (void)ws_size;
  const float*         x     = (const float*)d_in[0];
  const unsigned char* mask  = (const unsigned char*)d_in[1];
  const float*         w_qkv = (const float*)d_in[2];
  const float*         w_out = (const float*)d_in[3];
  const float*         b_out = (const float*)d_in[4];
  float* out = (float*)d_out;

  char* ws = (char*)d_ws;
  size_t off = 0;
  __bf16* xb    = (__bf16*)(ws + off); off += (size_t)B_ * N_ * DIM_ * 2;     // 8 MiB
  __bf16* wqkvT = (__bf16*)(ws + off); off += (size_t)QKVN_ * DIM_ * 2;       // 1.5 MiB
  __bf16* woutT = (__bf16*)(ws + off); off += (size_t)DIM_ * DIM_ * 2;        // 0.5 MiB
  __bf16* qd    = (__bf16*)(ws + off); off += (size_t)BH_ * N_ * DH_ * 2;     // 8 MiB
  __bf16* kd    = (__bf16*)(ws + off); off += (size_t)BH_ * N_ * DH_ * 2;     // 8 MiB
  __bf16* vtd   = (__bf16*)(ws + off); off += (size_t)BH_ * DH_ * N_ * 2;     // 8 MiB
  __bf16* ao    = (__bf16*)(ws + off); off += (size_t)B_ * N_ * H_ * DH_ * 2; // 8 MiB

  const int nx = B_ * N_ * DIM_;
  cvt_bf16_kernel<<<(nx + 255) / 256, 256, 0, stream>>>(x, xb, nx);
  cvt_transpose_kernel<<<(DIM_ * QKVN_ + 255) / 256, 256, 0, stream>>>(
      w_qkv, wqkvT, DIM_, QKVN_);
  cvt_transpose_kernel<<<(DIM_ * DIM_ + 255) / 256, 256, 0, stream>>>(
      w_out, woutT, DIM_, DIM_);

  qkv_gemm_kernel<<<dim3(QKVN_ / 128, (B_ * N_) / 64), 128, 0, stream>>>(
      xb, wqkvT, qd, kd, vtd);
  flash_attn_kernel<<<dim3(N_ / 128, BH_), 128, 0, stream>>>(
      qd, kd, vtd, mask, ao);
  out_gemm_kernel<<<dim3(DIM_ / 128, (B_ * N_) / 64), 128, 0, stream>>>(
      ao, woutT, b_out, out);
}
